// GCN_47236050322064
// MI455X (gfx1250) — matrix operand order repaired and checked by reading, not compiled, not run on hardware
//
#include <hip/hip_runtime.h>
#include <hip/hip_bf16.h>

// ---------------------------------------------------------------------------
// GAT pipeline for MI455X (gfx1250), wave32 + WMMA bf16.
// N=4096 nodes, NFEAT=256, NHID=32, H=8, NOUT=128.  All matmuls (dense GEMMs
// and the att@Wh aggregation) run on v_wmma_f32_16x16x32_bf16 with f32 acc.
// Two-pass softmax: exact row max/sum first, then a fused aggregation kernel
// with a ping-pong probability tile in LDS (exp work overlaps the WMMA pipe).
// ---------------------------------------------------------------------------

typedef __bf16 bf16_t;
typedef __attribute__((ext_vector_type(16))) __bf16 v16bf;
typedef __attribute__((ext_vector_type(8)))  __bf16 v8bf;
typedef __attribute__((ext_vector_type(8)))  float  v8f;

#define NN 4096
#define FF 256
#define ALPHA 0.2f

// Build WMMA A fragment (16x32 bf16, contiguous k = 16*g + s order) from a row-major row.
// rowk points at (row, k0); g = lane/16.
// VGPR v holds K halves {2*v + g*16, +1}.
__device__ __forceinline__ v16bf frag_a(const bf16_t* rowk, int g) {
  const v8bf* q = (const v8bf*)rowk;
  v8bf lo = q[2 * g];       // halves g*16   .. g*16+7
  v8bf hi = q[2 * g + 1];   // halves g*16+8 .. g*16+15
  v16bf r;
#pragma unroll
  for (int i = 0; i < 8; ++i) { r[i] = lo[i]; r[i + 8] = hi[i]; }
  return r;
}

// ---------------------------------------------------------------------------
// Dense GEMM: C[M,Nc] (f32) = A[M,K](bf16, row-major) @ BT[Nc,K]^T.
// 4 waves / block, each wave owns one 16x16 C tile (4 adjacent N tiles share
// the A rows through the WGP cache). ep: 0 = none, 1 = bias + relu.
// ---------------------------------------------------------------------------
__global__ void __launch_bounds__(128)
gemm_bf16_kernel(const bf16_t* __restrict__ A, const bf16_t* __restrict__ BT,
                 const float* __restrict__ bias, float* __restrict__ C,
                 int M, int Nc, int K, int ep) {
  const int lane = threadIdx.x & 31;
  const int wave = threadIdx.x >> 5;
  const int g = lane >> 4;
  const int ln = lane & 15;
  const int mt = blockIdx.y;
  const int nt = blockIdx.x * 4 + wave;
  const bf16_t* arow = A  + (size_t)(mt * 16 + ln) * K;
  const bf16_t* brow = BT + (size_t)(nt * 16 + ln) * K;
  v8f acc;
#pragma unroll
  for (int e = 0; e < 8; ++e) acc[e] = 0.f;
  for (int k0 = 0; k0 < K; k0 += 32) {
    v16bf a = frag_a(arow + k0, g);
    v16bf b = *(const v16bf*)(brow + k0 + g * 16);  // B: lane=col, K contiguous
    acc = __builtin_amdgcn_wmma_f32_16x16x32_bf16(false, a, false, b,
                                                  (short)0, acc, false, false);
  }
  const int col = nt * 16 + ln;
  const float bv = (ep == 1) ? bias[col] : 0.f;
#pragma unroll
  for (int r = 0; r < 8; ++r) {       // C layout: VGPR r -> row r + 8*g
    int row = mt * 16 + r + (g << 3);
    float v = acc[r];
    if (ep == 1) { v += bv; v = fmaxf(v, 0.f); }
    C[(size_t)row * Nc + col] = v;
  }
}

// ---------------------------------------------------------------------------
// Fused GAT aggregation, templated on head count (HH = 8 or 1, HH*Fo == 256):
//   out[n, h*Fo+o] = elu( (1/s_hn) * sum_j p_hnj * Wh[j, h*Fo+o] )
// Block = 128 threads (4 waves); each wave owns 4 of the 16 feature col-tiles.
// Probability producer: each thread owns a fixed (head,row) slice, so es/m and
// the adjacency word live in registers; one bitmask word load per 32-step.
// LDS tile is ping-ponged: produce p(j0+32) while WMMA consumes p(j0);
// a single s_barrier per step.
// ---------------------------------------------------------------------------
template <int HH>
__global__ void __launch_bounds__(128)
gat_agg_kernel(const bf16_t* __restrict__ WhT,   // [256, N] bf16 (feature-major)
               const float* __restrict__ es, const float* __restrict__ ed,
               const float* __restrict__ mrow, const float* __restrict__ rrow,
               const unsigned* __restrict__ bits,  // [N, N/32]
               float* __restrict__ out,            // [N, 256] f32
               int N) {
  constexpr int FO   = 256 / HH;       // features per head
  constexpr int ROWS = HH * 16;        // (head,row) pairs in the p tile
  constexpr int SL   = 128 / ROWS;     // j-slices per row
  constexpr int CW   = 32 / SL;        // neighbor columns per thread
  __shared__ __align__(32) bf16_t p_lds[2 * 4096]; // ping-pong, HH*16*32 used

  const int n0 = blockIdx.x * 16;
  const int tid = threadIdx.x;
  const int lane = tid & 31;
  const int wave = tid >> 5;
  const int g = lane >> 4;
  const int ln = lane & 15;

  // ---- producer role (fixed for the whole kernel) ----
  const int rid = tid % ROWS;          // (head,row) pair
  const int ph = rid >> 4;
  const int pr = rid & 15;
  const int jb = (tid / ROWS) * CW;    // first neighbor col in this slice
  const int pnode = n0 + pr;
  const float esn = es[ph * N + pnode];
  const float mn  = mrow[ph * N + pnode];
  const unsigned* __restrict__ pbits = bits + (size_t)pnode * (N >> 5);
  const float* __restrict__ edh = ed + ph * N;
  const int pofs = (ph << 9) + (pr << 5) + jb;  // element offset inside buffer

  // ---- consumer role: 4 col tiles per wave ----
  v8f acc[4];
#pragma unroll
  for (int i = 0; i < 4; ++i)
#pragma unroll
    for (int e = 0; e < 8; ++e) acc[i][e] = 0.f;
  const bf16_t* brow[4];
  int bhead[4];
#pragma unroll
  for (int i = 0; i < 4; ++i) {
    int ct = wave * 4 + i;
    brow[i] = WhT + (size_t)(ct * 16 + ln) * N;
    bhead[i] = (ct * 16) / FO;
  }

  // produce first tile (j0 = 0) into buffer 0
  {
    unsigned w = pbits[0];
#pragma unroll
    for (int jj = 0; jj < CW; ++jj) {
      int j = jb + jj;
      float pv = 0.f;
      if ((w >> j) & 1u) {
        float e = esn + edh[j];
        e = e > 0.f ? e : ALPHA * e;          // leaky relu
        pv = __expf(e - mn);                  // exact row max known
      }
      p_lds[pofs + jj] = (bf16_t)pv;
    }
  }
  __syncthreads();

  const int steps = N >> 5;
  for (int it = 0; it < steps; ++it) {
    const int j0 = it << 5;
    const int cur = (it & 1) << 12;           // 4096-element buffers
    const int nxt = cur ^ 4096;
    if (it + 1 < steps) {                     // produce next tile (overlaps WMMA)
      const int j0n = j0 + 32;
      unsigned w = pbits[j0n >> 5];
#pragma unroll
      for (int jj = 0; jj < CW; ++jj) {
        int j = jb + jj;
        float pv = 0.f;
        if ((w >> j) & 1u) {
          float e = esn + edh[j0n + j];
          e = e > 0.f ? e : ALPHA * e;
          pv = __expf(e - mn);
        }
        p_lds[nxt + pofs + jj] = (bf16_t)pv;
      }
    }
#pragma unroll
    for (int i = 0; i < 4; ++i) {             // consume current tile
      const int base = cur + (bhead[i] << 9) + ln * 32;
      v16bf a;
#pragma unroll
      for (int u = 0; u < 8; ++u) {           // A-layout dword gather from LDS
        int d = u + (g << 3);
        a[2 * u]     = p_lds[base + 2 * d];
        a[2 * u + 1] = p_lds[base + 2 * d + 1];
      }
      v16bf b = *(const v16bf*)(brow[i] + j0 + g * 16);
      acc[i] = __builtin_amdgcn_wmma_f32_16x16x32_bf16(false, a, false, b,
                                                       (short)0, acc[i], false, false);
    }
    __syncthreads();                          // single barrier per step
  }

#pragma unroll
  for (int i = 0; i < 4; ++i) {
    int ct = wave * 4 + i;
    int h = bhead[i];
    int col = ct * 16 + ln;
#pragma unroll
    for (int r = 0; r < 8; ++r) {
      int rr = r + (g << 3);
      float v = acc[i][r] * rrow[h * N + n0 + rr];   // 1/sum
      v = v > 0.f ? v : (__expf(v) - 1.f);           // elu
      out[(size_t)(n0 + rr) * 256 + col] = v;
    }
  }
}

// ---------------------------------------------------------------------------
// Softmax statistics: exact row max + sum of exp over masked neighbors.
// One wave per (head, node); bitmask word broadcast per iteration.
// ---------------------------------------------------------------------------
__global__ void stats_kernel(const float* __restrict__ es, const float* __restrict__ ed,
                             const unsigned* __restrict__ bits,
                             float* __restrict__ mrow, float* __restrict__ rrow,
                             int N, int H) {
  int gtid = blockIdx.x * blockDim.x + threadIdx.x;
  int wid = gtid >> 5;
  int lane = gtid & 31;
  if (wid >= N * H) return;
  int h = wid / N;
  int n = wid - h * N;
  float esn = es[h * N + n];
  const unsigned* brw = bits + (size_t)n * (N >> 5);
  const float* edh = ed + h * N;
  float m = -3.0e38f;
  for (int k = 0; k < (N >> 5); ++k) {
    unsigned w = brw[k];
    if ((w >> lane) & 1u) {
      float e = esn + edh[(k << 5) + lane];
      e = e > 0.f ? e : ALPHA * e;
      m = fmaxf(m, e);
    }
  }
#pragma unroll
  for (int off = 16; off > 0; off >>= 1) m = fmaxf(m, __shfl_xor(m, off, 32));
  float s = 0.f;
  for (int k = 0; k < (N >> 5); ++k) {
    unsigned w = brw[k];
    if ((w >> lane) & 1u) {
      float e = esn + edh[(k << 5) + lane];
      e = e > 0.f ? e : ALPHA * e;
      s += __expf(e - m);
    }
  }
#pragma unroll
  for (int off = 16; off > 0; off >>= 1) s += __shfl_xor(s, off, 32);
  if (lane == 0) { mrow[h * N + n] = m; rrow[h * N + n] = 1.f / s; }
}

// es[h][n] = Wh[n, h*Fo:(h+1)*Fo] . a_s[h];  ed likewise. Thread per (n,h).
__global__ void scores_kernel(const float* __restrict__ Wh, const float* __restrict__ as,
                              const float* __restrict__ ad, float* __restrict__ es,
                              float* __restrict__ ed, int N, int H, int Fo) {
  int t = blockIdx.x * blockDim.x + threadIdx.x;
  if (t >= N * H) return;
  int n = t / H, h = t - n * H;
  const float* row = Wh + (size_t)n * (H * Fo) + h * Fo;
  float s = 0.f, d = 0.f;
  for (int o = 0; o < Fo; ++o) {
    float w = row[o];
    s += w * as[h * Fo + o];
    d += w * ad[h * Fo + o];
  }
  es[h * N + n] = s;
  ed[h * N + n] = d;
}

// f32 -> bf16 (row-major copy)
__global__ void cvt_bf16_kernel(const float* __restrict__ s, bf16_t* __restrict__ d, int n) {
  int i = blockIdx.x * blockDim.x + threadIdx.x;
  if (i < n) d[i] = (bf16_t)s[i];
}

// dst[c*R + r] = (bf16) src[r*C + c]   (transpose-convert, coalesced writes)
__global__ void cvt_t_bf16_kernel(const float* __restrict__ s, bf16_t* __restrict__ d,
                                  int R, int C) {
  int i = blockIdx.x * blockDim.x + threadIdx.x;
  if (i >= R * C) return;
  int c = i / R, r = i - c * R;
  d[i] = (bf16_t)s[(size_t)r * C + c];
}

// WT[(h*Fo+o)*Fin + f] = (bf16) W[(h*Fin+f)*Fo + o]   (per-head weight transpose)
__global__ void pack_whead_kernel(const float* __restrict__ W, bf16_t* __restrict__ WT,
                                  int H, int Fin, int Fo) {
  int t = blockIdx.x * blockDim.x + threadIdx.x;
  if (t >= H * Fin * Fo) return;
  int rowOut = t / Fin, f = t - rowOut * Fin;
  int h = rowOut / Fo, o = rowOut - h * Fo;
  WT[t] = (bf16_t)W[((size_t)(h * Fin + f)) * Fo + o];
}

// adjacency f32 -> bitmask u32 [N, N/32]
__global__ void pack_adj_kernel(const float* __restrict__ adj, unsigned* __restrict__ bits,
                                int N) {
  int t = blockIdx.x * blockDim.x + threadIdx.x;
  int words = N >> 5;
  if (t >= N * words) return;
  int n = t / words, w = t - n * words;
  const float* row = adj + (size_t)n * N + (w << 5);
  unsigned b = 0;
#pragma unroll
  for (int j = 0; j < 32; ++j) b |= (row[j] > 0.f) ? (1u << j) : 0u;
  bits[t] = b;
}

__global__ void mul_kernel(const float* __restrict__ a, const float* __restrict__ b,
                           float* __restrict__ o, int n) {
  int i = blockIdx.x * blockDim.x + threadIdx.x;
  if (i < n) o[i] = a[i] * b[i];
}

// ---------------------------------------------------------------------------
extern "C" void kernel_launch(void* const* d_in, const int* in_sizes, int n_in,
                              void* d_out, int out_size, void* d_ws, size_t ws_size,
                              hipStream_t stream) {
  (void)in_sizes; (void)n_in; (void)out_size; (void)ws_size;
  const float* x    = (const float*)d_in[0];
  const float* adj  = (const float*)d_in[1];
  const float* W1   = (const float*)d_in[2];
  const float* a1s  = (const float*)d_in[3];
  const float* a1d  = (const float*)d_in[4];
  const float* Wo1  = (const float*)d_in[5];
  const float* ao1s = (const float*)d_in[6];
  const float* ao1d = (const float*)d_in[7];
  const float* W2   = (const float*)d_in[8];
  const float* a2s  = (const float*)d_in[9];
  const float* a2d  = (const float*)d_in[10];
  const float* Wo2  = (const float*)d_in[11];
  const float* ao2s = (const float*)d_in[12];
  const float* ao2d = (const float*)d_in[13];
  const float* Wt   = (const float*)d_in[14];
  const float* bt   = (const float*)d_in[15];
  const float* Wf1  = (const float*)d_in[16];
  const float* bf1  = (const float*)d_in[17];
  const float* Wf2  = (const float*)d_in[18];
  const float* bf2  = (const float*)d_in[19];
  const float* Wl1  = (const float*)d_in[20];
  const float* bl1  = (const float*)d_in[21];
  const float* Wl2  = (const float*)d_in[22];
  const float* bl2  = (const float*)d_in[23];

  // ---- carve workspace (~22 MB) ----
  char* wp = (char*)d_ws;
  auto carve = [&](size_t bytes) -> void* {
    void* r = (void*)wp;
    wp += (bytes + 255) & ~(size_t)255;
    return r;
  };
  bf16_t* xbf   = (bf16_t*)carve((size_t)NN * FF * 2);   // bf16 activations (A)
  bf16_t* actA  = (bf16_t*)carve((size_t)NN * FF * 2);   // bf16 activations (A)
  bf16_t* actT  = (bf16_t*)carve((size_t)FF * NN * 2);   // bf16 WhT [256, N]
  float*  F1    = (float*) carve((size_t)NN * FF * 4);   // f32 scratch
  float*  F2    = (float*) carve((size_t)NN * FF * 4);   // f32 scratch
  float*  esB   = (float*) carve((size_t)8 * NN * 4);
  float*  edB   = (float*) carve((size_t)8 * NN * 4);
  float*  mB    = (float*) carve((size_t)8 * NN * 4);
  float*  rB    = (float*) carve((size_t)8 * NN * 4);
  unsigned* adjbits = (unsigned*)carve((size_t)NN * (NN / 32) * 4);
  bf16_t* WcatT1 = (bf16_t*)carve((size_t)FF * FF * 2);
  bf16_t* WoT1   = (bf16_t*)carve((size_t)FF * FF * 2);
  bf16_t* WcatT2 = (bf16_t*)carve((size_t)FF * FF * 2);
  bf16_t* WoT2   = (bf16_t*)carve((size_t)FF * FF * 2);
  bf16_t* WtT    = (bf16_t*)carve((size_t)FF * FF * 2);
  bf16_t* Wf1T   = (bf16_t*)carve((size_t)FF * FF * 2);
  bf16_t* Wf2T   = (bf16_t*)carve((size_t)128 * FF * 2);
  bf16_t* Wl1T   = (bf16_t*)carve((size_t)FF * FF * 2);
  bf16_t* Wl2T   = (bf16_t*)carve((size_t)128 * FF * 2);
  float*  D1     = (float*) carve((size_t)NN * 128 * 4);
  float*  D2     = (float*) carve((size_t)NN * 128 * 4);

  auto gemm = [&](const bf16_t* A, const bf16_t* BT, const float* bias, float* C,
                  int M, int Nc, int K, int ep) {
    dim3 grd(Nc / 64, M / 16);
    gemm_bf16_kernel<<<grd, 128, 0, stream>>>(A, BT, bias, C, M, Nc, K, ep);
  };
  auto cvt = [&](const float* s, bf16_t* d, int n) {
    cvt_bf16_kernel<<<(n + 255) / 256, 256, 0, stream>>>(s, d, n);
  };
  auto cvtT = [&](const float* s, bf16_t* d, int R, int C) {
    cvt_t_bf16_kernel<<<(R * C + 255) / 256, 256, 0, stream>>>(s, d, R, C);
  };
  auto packw = [&](const float* W, bf16_t* WT, int H, int Fin, int Fo) {
    pack_whead_kernel<<<(H * Fin * Fo + 255) / 256, 256, 0, stream>>>(W, WT, H, Fin, Fo);
  };
  // full attention layer: Wh = act @ WT^T ; scores ; stats ; aggregate (+elu)
  auto attn = [&](const bf16_t* actIn, const bf16_t* WT, const float* as,
                  const float* ad, int H, int Fo, float* outF) {
    gemm(actIn, WT, nullptr, F1, NN, FF, FF, 0);
    scores_kernel<<<(NN * H + 255) / 256, 256, 0, stream>>>(F1, as, ad, esB, edB, NN, H, Fo);
    cvtT(F1, actT, NN, FF);
    stats_kernel<<<(NN * H) / 8, 256, 0, stream>>>(esB, edB, adjbits, mB, rB, NN, H);
    if (H == 8)
      gat_agg_kernel<8><<<NN / 16, 128, 0, stream>>>(actT, esB, edB, mB, rB,
                                                     adjbits, outF, NN);
    else
      gat_agg_kernel<1><<<NN / 16, 128, 0, stream>>>(actT, esB, edB, mB, rB,
                                                     adjbits, outF, NN);
  };

  // ---- prep ----
  cvt(x, xbf, NN * FF);
  packw(W1, WcatT1, 8, FF, 32);
  packw(Wo1, WoT1, 1, FF, FF);
  packw(W2, WcatT2, 8, FF, 32);
  packw(Wo2, WoT2, 1, FF, FF);
  packw(Wt, WtT, 1, FF, FF);
  packw(Wf1, Wf1T, 1, FF, FF);
  packw(Wf2, Wf2T, 1, FF, 128);
  packw(Wl1, Wl1T, 1, FF, FF);
  packw(Wl2, Wl2T, 1, FF, 128);
  pack_adj_kernel<<<(NN * (NN / 32) + 255) / 256, 256, 0, stream>>>(adj, adjbits, NN);

  // ---- GraphAtten block 1 ----
  attn(xbf, WcatT1, a1s, a1d, 8, 32, F2);  cvt(F2, actA, NN * FF);
  attn(actA, WoT1, ao1s, ao1d, 1, FF, F2); cvt(F2, actA, NN * FF);
  // ---- GraphAtten block 2 ----
  attn(actA, WcatT2, a2s, a2d, 8, 32, F2); cvt(F2, actA, NN * FF);
  attn(actA, WoT2, ao2s, ao2d, 1, FF, F2); cvt(F2, actA, NN * FF);

  // ---- tran1 + dual-branch sparsification ----
  gemm(actA, WtT, bt, F1, NN, FF, FF, 1);  cvt(F1, actA, NN * FF);  // relu(h@Wt+bt)
  gemm(actA, Wf1T, bf1, F2, NN, FF, FF, 1); cvt(F2, xbf, NN * FF);
  gemm(xbf, Wf2T, bf2, D1, NN, 128, FF, 1);                          // desc
  gemm(actA, Wl1T, bl1, F2, NN, FF, FF, 1); cvt(F2, xbf, NN * FF);
  gemm(xbf, Wl2T, bl2, D2, NN, 128, FF, 1);                          // loc
  mul_kernel<<<(NN * 128 + 255) / 256, 256, 0, stream>>>(D1, D2, (float*)d_out, NN * 128);
}
